// Mamba2GestureRecognizer_47811575939625
// MI455X (gfx1250) — compile-verified, matrix operand order checked by reading, not run on hardware
//
#include <hip/hip_runtime.h>
#include <hip/hip_bf16.h>
#include <math.h>

// ---------------- model constants ----------------
static constexpr int D_IN    = 63;
static constexpr int D_MODEL = 256;
static constexpr int N_CLS   = 14;
static constexpr int NL      = 4;
static constexpr int D_INNER = 512;
static constexpr int D_STATE = 128;
static constexpr int NH      = 8;
static constexpr int HP      = 64;
static constexpr int DCONV   = 4;
static constexpr int CONV_CH = D_INNER + 2 * D_STATE;           // 768
static constexpr int PROJ    = 2 * D_INNER + 2 * D_STATE + NH;  // 1288
static constexpr int PROJ_P  = 1344;                            // padded to mult of 64
static constexpr int Bsz     = 32;
static constexpr int Lsz     = 1024;
static constexpr int Ttok    = Bsz * Lsz;                       // 32768
static constexpr float EPS   = 1e-5f;
static constexpr int SCHUNK  = 8;                               // scan steps per LDS chunk

// ---------------- WMMA types ----------------
typedef __bf16 bf16;
typedef __attribute__((ext_vector_type(16))) __bf16 v16bf;
typedef __attribute__((ext_vector_type(8)))  __bf16 v8bf;
typedef __attribute__((ext_vector_type(8)))  float  v8f;

__device__ __forceinline__ float sigmoidf_(float x) { return 1.0f / (1.0f + __expf(-x)); }
__device__ __forceinline__ float siluf_(float x)    { return x * sigmoidf_(x); }
__device__ __forceinline__ float softplusf_(float x){ return (x > 20.0f) ? x : log1pf(__expf(x)); }

__device__ __forceinline__ float blockReduceSum(float v, float* sdata, int tid, int nthreads) {
    sdata[tid] = v;
    __syncthreads();
    for (int s = nthreads >> 1; s > 0; s >>= 1) {
        if (tid < s) sdata[tid] += sdata[tid + s];
        __syncthreads();
    }
    float r = sdata[0];
    __syncthreads();
    return r;
}

__device__ __forceinline__ v16bf load_frag(const bf16* p) {
    v8bf lo = *(const v8bf*)(p);
    v8bf hi = *(const v8bf*)(p + 16);
    return __builtin_shufflevector(lo, hi, 0,1,2,3,4,5,6,7,8,9,10,11,12,13,14,15);
}

// ---------------- weight conversion ----------------
// inproj (NL,1288,256) f32 -> (NL,1344,256) bf16, zero-padded rows
__global__ void k_cvt_inproj(const float* __restrict__ src, bf16* __restrict__ dst) {
    int idx = blockIdx.x * 256 + threadIdx.x;
    int total = NL * PROJ_P * D_MODEL;
    if (idx >= total) return;
    int layer = idx / (PROJ_P * D_MODEL);
    int rem   = idx % (PROJ_P * D_MODEL);
    int r     = rem / D_MODEL;
    int c     = rem % D_MODEL;
    float v = (r < PROJ) ? src[(size_t)layer * PROJ * D_MODEL + (size_t)r * D_MODEL + c] : 0.0f;
    dst[idx] = (bf16)v;
}

// outp (NL,256,512) f32 -> bf16 straight
__global__ void k_cvt_outp(const float* __restrict__ src, bf16* __restrict__ dst) {
    int idx = blockIdx.x * 256 + threadIdx.x;
    int total = NL * D_MODEL * D_INNER;
    if (idx >= total) return;
    dst[idx] = (bf16)src[idx];
}

// ---------------- embed + input LayerNorm ----------------
__global__ void k_embed_ln(const float* __restrict__ x, const float* __restrict__ in_w,
                           const float* __restrict__ in_b, const float* __restrict__ lnw,
                           const float* __restrict__ lnb, float* __restrict__ hF,
                           bf16* __restrict__ hB) {
    __shared__ float sdata[D_MODEL];
    int t = blockIdx.x;
    int d = threadIdx.x;
    const float* xr = x + (size_t)t * D_IN;
    const float* wr = in_w + (size_t)d * D_IN;
    float acc = in_b[d];
    #pragma unroll
    for (int j = 0; j < D_IN; ++j) acc += xr[j] * wr[j];
    float mu = blockReduceSum(acc, sdata, d, D_MODEL) * (1.0f / D_MODEL);
    float c  = acc - mu;
    float var = blockReduceSum(c * c, sdata, d, D_MODEL) * (1.0f / D_MODEL);
    float h = c * rsqrtf(var + EPS) * lnw[d] + lnb[d];
    hF[(size_t)t * D_MODEL + d] = h;
    hB[(size_t)t * D_MODEL + d] = (bf16)h;
}

// ---------------- bf16 WMMA GEMM: C(MxN) = A(MxK,bf16) * W(NxK,bf16)^T ----------------
// block = 128 threads (4 waves); block tile 128x64; wave tile 32x64 (2x4 WMMA tiles).
// M % 128 == 0, N % 64 == 0, K % 32 == 0 guaranteed by padding.
__global__ void k_gemm_bf16(const bf16* __restrict__ A, const bf16* __restrict__ W,
                            float* __restrict__ C, int M, int N, int K) {
    const int wave = threadIdx.x >> 5;
    const int lane = threadIdx.x & 31;
    const int m16  = lane & 15;
    const int kh   = lane >> 4;     // which K-half this lane covers

    const int row0 = blockIdx.y * 128 + wave * 32;
    const int col0 = blockIdx.x * 64;

    v8f acc00 = {}, acc01 = {}, acc02 = {}, acc03 = {};
    v8f acc10 = {}, acc11 = {}, acc12 = {}, acc13 = {};

    const size_t aRow0 = (size_t)(row0 +  0 + m16) * K;
    const size_t aRow1 = (size_t)(row0 + 16 + m16) * K;
    const size_t bRow0 = (size_t)(col0 +  0 + m16) * K;
    const size_t bRow1 = (size_t)(col0 + 16 + m16) * K;
    const size_t bRow2 = (size_t)(col0 + 32 + m16) * K;
    const size_t bRow3 = (size_t)(col0 + 48 + m16) * K;

    for (int k0 = 0; k0 < K; k0 += 32) {
        const int ko = k0 + kh * 8;
        v16bf a0 = load_frag(A + aRow0 + ko);
        v16bf a1 = load_frag(A + aRow1 + ko);
        v16bf b0 = load_frag(W + bRow0 + ko);
        v16bf b1 = load_frag(W + bRow1 + ko);
        v16bf b2 = load_frag(W + bRow2 + ko);
        v16bf b3 = load_frag(W + bRow3 + ko);

        acc00 = __builtin_amdgcn_wmma_f32_16x16x32_bf16(false, a0, false, b0, (short)0, acc00, false, false);
        acc01 = __builtin_amdgcn_wmma_f32_16x16x32_bf16(false, a0, false, b1, (short)0, acc01, false, false);
        acc02 = __builtin_amdgcn_wmma_f32_16x16x32_bf16(false, a0, false, b2, (short)0, acc02, false, false);
        acc03 = __builtin_amdgcn_wmma_f32_16x16x32_bf16(false, a0, false, b3, (short)0, acc03, false, false);
        acc10 = __builtin_amdgcn_wmma_f32_16x16x32_bf16(false, a1, false, b0, (short)0, acc10, false, false);
        acc11 = __builtin_amdgcn_wmma_f32_16x16x32_bf16(false, a1, false, b1, (short)0, acc11, false, false);
        acc12 = __builtin_amdgcn_wmma_f32_16x16x32_bf16(false, a1, false, b2, (short)0, acc12, false, false);
        acc13 = __builtin_amdgcn_wmma_f32_16x16x32_bf16(false, a1, false, b3, (short)0, acc13, false, false);
    }

    // D layout: VGPR r -> row (r + 8*kh), lane%16 -> column
    #pragma unroll
    for (int r = 0; r < 8; ++r) {
        size_t rowA = (size_t)(row0 +  0 + 8 * kh + r) * N;
        C[rowA + col0 +  0 + m16] = acc00[r];
        C[rowA + col0 + 16 + m16] = acc01[r];
        C[rowA + col0 + 32 + m16] = acc02[r];
        C[rowA + col0 + 48 + m16] = acc03[r];
        size_t rowB = (size_t)(row0 + 16 + 8 * kh + r) * N;
        C[rowB + col0 +  0 + m16] = acc10[r];
        C[rowB + col0 + 16 + m16] = acc11[r];
        C[rowB + col0 + 32 + m16] = acc12[r];
        C[rowB + col0 + 48 + m16] = acc13[r];
    }
}

// ---------------- depthwise causal conv (window 4) + SiLU ----------------
__global__ void k_conv_silu(const float* __restrict__ zx, const float* __restrict__ cw,
                            const float* __restrict__ cb, float* __restrict__ xact) {
    int t = blockIdx.x;
    int c = blockIdx.y * 256 + threadIdx.x;       // 0..767
    int b = t >> 10;
    int l = t & 1023;
    float acc = cb[c];
    #pragma unroll
    for (int k = 0; k < DCONV; ++k) {
        int lp = l + k - (DCONV - 1);
        if (lp >= 0)
            acc += zx[((size_t)b * Lsz + lp) * PROJ_P + D_INNER + c] * cw[k * CONV_CH + c];
    }
    xact[(size_t)t * CONV_CH + c] = siluf_(acc);
}

// ---------------- selective scan (chunked) ----------------
// one block per (b, head); 256 threads; state h[64][128] in regs (32 f32/thread)
// thread tid: p = tid & 63, n-range = (tid>>6)*32 .. +32
// LDS tile layout per step: [0..63]=x, [64..191]=B, [192..319]=C
__global__ void k_scan(const float* __restrict__ zx, const float* __restrict__ xact,
                       const float* __restrict__ dt_bias, const float* __restrict__ A_log,
                       const float* __restrict__ Dp, float* __restrict__ ys) {
    __shared__ float tile[SCHUNK][320];
    __shared__ float ypart[SCHUNK][256];
    __shared__ float dts[SCHUNK][2];

    const int tid = threadIdx.x;
    const int b  = blockIdx.x >> 3;
    const int hh = blockIdx.x & 7;
    const int p  = tid & 63;
    const int nb = (tid >> 6) * 32;

    const float dtb = dt_bias[hh];
    const float Ah  = -__expf(A_log[hh]);
    const float Dh  = Dp[hh];

    float s[32];
    #pragma unroll
    for (int j = 0; j < 32; ++j) s[j] = 0.0f;

    for (int tl0 = 0; tl0 < Lsz; tl0 += SCHUNK) {
        const size_t row0 = (size_t)b * Lsz + tl0;

        // ---- load phase: 8 steps of (x, B, C) + dt/dA ----
        #pragma unroll
        for (int idx = tid; idx < SCHUNK * 320; idx += 256) {
            int st = idx / 320;
            int o  = idx - st * 320;
            int col = (o < 64) ? hh * HP + o : 448 + o;   // B at 512+, C at 640+
            tile[st][o] = xact[(row0 + st) * CONV_CH + col];
        }
        if (tid < SCHUNK) {
            float raw = zx[(row0 + tid) * PROJ_P + (D_INNER + CONV_CH) + hh];
            float dt  = softplusf_(raw + dtb);
            dts[tid][0] = dt;
            dts[tid][1] = __expf(dt * Ah);
        }
        __syncthreads();

        // ---- recurrence: 8 sequential steps out of LDS/registers ----
        #pragma unroll
        for (int st = 0; st < SCHUNK; ++st) {
            const float dt = dts[st][0];
            const float dA = dts[st][1];
            const float c1 = dt * tile[st][p];
            float yl = (nb == 0) ? Dh * tile[st][p] : 0.0f;  // fold D*x into group 0
            #pragma unroll
            for (int j = 0; j < 32; ++j) {
                s[j] = s[j] * dA + c1 * tile[st][64 + nb + j];
                yl  += s[j] * tile[st][192 + nb + j];
            }
            ypart[st][tid] = yl;
        }
        __syncthreads();

        // ---- reduce 4 n-groups and store y for 8 steps ----
        #pragma unroll
        for (int ss = 0; ss < 2; ++ss) {
            int st = (tid >> 6) + ss * 4;
            float y = ypart[st][p] + ypart[st][p + 64] + ypart[st][p + 128] + ypart[st][p + 192];
            ys[(row0 + st) * D_INNER + hh * HP + p] = y;
        }
        // next-chunk tile writes are fenced by the first __syncthreads of the next chunk
    }
}

// ---------------- gate (silu(z)) + RMSNorm -> bf16 ----------------
__global__ void k_gate_rms(const float* __restrict__ ys, const float* __restrict__ zx,
                           const float* __restrict__ normw, bf16* __restrict__ gB) {
    __shared__ float sdata[D_INNER];
    int t = blockIdx.x;
    int d = threadIdx.x;
    float y = ys[(size_t)t * D_INNER + d];
    float z = zx[(size_t)t * PROJ_P + d];
    float g = y * siluf_(z);
    float ms = blockReduceSum(g * g, sdata, d, D_INNER) * (1.0f / D_INNER);
    float out = g * rsqrtf(ms + EPS) * normw[d];
    gB[(size_t)t * D_INNER + d] = (bf16)out;
}

// ---------------- residual add + LayerNorm -> hF (f32) + hB (bf16) ----------------
__global__ void k_add_ln(const float* __restrict__ mF, float* __restrict__ hF,
                         bf16* __restrict__ hB, const float* __restrict__ lnw,
                         const float* __restrict__ lnb) {
    __shared__ float sdata[D_MODEL];
    int t = blockIdx.x;
    int d = threadIdx.x;
    float v = mF[(size_t)t * D_MODEL + d] + hF[(size_t)t * D_MODEL + d];
    float mu = blockReduceSum(v, sdata, d, D_MODEL) * (1.0f / D_MODEL);
    float c = v - mu;
    float var = blockReduceSum(c * c, sdata, d, D_MODEL) * (1.0f / D_MODEL);
    float h = c * rsqrtf(var + EPS) * lnw[d] + lnb[d];
    hF[(size_t)t * D_MODEL + d] = h;
    hB[(size_t)t * D_MODEL + d] = (bf16)h;
}

// ---------------- masked mean pool ----------------
__global__ void k_pool(const float* __restrict__ hF, const int* __restrict__ lengths,
                       float* __restrict__ pooled) {
    int b = blockIdx.x;
    int d = threadIdx.x;
    int len = lengths[b];
    float acc = 0.0f;
    for (int l = 0; l < len; ++l) acc += hF[((size_t)b * Lsz + l) * D_MODEL + d];
    pooled[b * D_MODEL + d] = acc / (float)len;
}

// ---------------- classifier head ----------------
__global__ void k_head(const float* __restrict__ pooled, const float* __restrict__ hw,
                       const float* __restrict__ hb, float* __restrict__ out) {
    int b = blockIdx.x;
    int c = threadIdx.x;
    if (c >= N_CLS) return;
    const float* pr = pooled + b * D_MODEL;
    const float* wr = hw + c * D_MODEL;
    float acc = hb[c];
    for (int d = 0; d < D_MODEL; ++d) acc += pr[d] * wr[d];
    out[b * N_CLS + c] = acc;
}

// ---------------- launcher ----------------
extern "C" void kernel_launch(void* const* d_in, const int* in_sizes, int n_in,
                              void* d_out, int out_size, void* d_ws, size_t ws_size,
                              hipStream_t stream) {
    const float* x        = (const float*)d_in[0];
    const float* in_w     = (const float*)d_in[1];
    const float* in_b     = (const float*)d_in[2];
    const float* lnin_w   = (const float*)d_in[3];
    const float* lnin_b   = (const float*)d_in[4];
    const float* inproj_w = (const float*)d_in[5];
    const float* conv_w   = (const float*)d_in[6];
    const float* conv_b   = (const float*)d_in[7];
    const float* dt_bias  = (const float*)d_in[8];
    const float* A_log    = (const float*)d_in[9];
    const float* Dparam   = (const float*)d_in[10];
    const float* norm_w   = (const float*)d_in[11];
    const float* outp_w   = (const float*)d_in[12];
    const float* ln_w     = (const float*)d_in[13];
    const float* ln_b     = (const float*)d_in[14];
    const float* head_w   = (const float*)d_in[15];
    const float* head_b   = (const float*)d_in[16];
    const int*   lengths  = (const int*)d_in[17];
    float* out = (float*)d_out;

    // workspace carve-up (256B aligned)
    char* base = (char*)d_ws;
    size_t off = 0;
    auto carve = [&](size_t bytes) -> void* {
        off = (off + 255) & ~(size_t)255;
        void* p = base + off;
        off += bytes;
        return p;
    };
    float* hF    = (float*)carve((size_t)Ttok * D_MODEL * 4);
    bf16*  hB    = (bf16*) carve((size_t)Ttok * D_MODEL * 2);
    bf16*  WinB  = (bf16*) carve((size_t)NL * PROJ_P * D_MODEL * 2);
    bf16*  WoutB = (bf16*) carve((size_t)NL * D_MODEL * D_INNER * 2);
    float* zx    = (float*)carve((size_t)Ttok * PROJ_P * 4);
    float* xact  = (float*)carve((size_t)Ttok * CONV_CH * 4);
    float* ys    = (float*)carve((size_t)Ttok * D_INNER * 4);
    bf16*  gB    = (bf16*) carve((size_t)Ttok * D_INNER * 2);
    float* mF    = (float*)carve((size_t)Ttok * D_MODEL * 4);
    float* pooled= (float*)carve((size_t)Bsz * D_MODEL * 4);
    (void)ws_size; (void)n_in; (void)in_sizes; (void)out_size;

    // weight conversion
    {
        int total = NL * PROJ_P * D_MODEL;
        k_cvt_inproj<<<(total + 255) / 256, 256, 0, stream>>>(inproj_w, WinB);
        total = NL * D_MODEL * D_INNER;
        k_cvt_outp<<<(total + 255) / 256, 256, 0, stream>>>(outp_w, WoutB);
    }

    // embed + input LN
    k_embed_ln<<<Ttok, D_MODEL, 0, stream>>>(x, in_w, in_b, lnin_w, lnin_b, hF, hB);

    for (int i = 0; i < NL; ++i) {
        const bf16*  Win_l  = WinB  + (size_t)i * PROJ_P * D_MODEL;
        const bf16*  Wout_l = WoutB + (size_t)i * D_MODEL * D_INNER;
        const float* cw_l   = conv_w  + (size_t)i * DCONV * CONV_CH;
        const float* cb_l   = conv_b  + (size_t)i * CONV_CH;
        const float* dtb_l  = dt_bias + (size_t)i * NH;
        const float* Al_l   = A_log   + (size_t)i * NH;
        const float* D_l    = Dparam  + (size_t)i * NH;
        const float* nw_l   = norm_w  + (size_t)i * D_INNER;
        const float* lnw_l  = ln_w    + (size_t)i * D_MODEL;
        const float* lnb_l  = ln_b    + (size_t)i * D_MODEL;

        // in_proj: zx(T x 1344) = hB(T x 256) * Win^T
        {
            dim3 grid(PROJ_P / 64, Ttok / 128);
            k_gemm_bf16<<<grid, 128, 0, stream>>>(hB, Win_l, zx, Ttok, PROJ_P, D_MODEL);
        }
        // conv + silu
        {
            dim3 grid(Ttok, CONV_CH / 256);
            k_conv_silu<<<grid, 256, 0, stream>>>(zx, cw_l, cb_l, xact);
        }
        // selective scan
        k_scan<<<Bsz * NH, 256, 0, stream>>>(zx, xact, dtb_l, Al_l, D_l, ys);
        // gate + RMSNorm -> bf16
        k_gate_rms<<<Ttok, D_INNER, 0, stream>>>(ys, zx, nw_l, gB);
        // out_proj: mF(T x 256) = gB(T x 512) * Wout^T
        {
            dim3 grid(D_MODEL / 64, Ttok / 128);
            k_gemm_bf16<<<grid, 128, 0, stream>>>(gB, Wout_l, mF, Ttok, D_MODEL, D_INNER);
        }
        // residual + LN
        k_add_ln<<<Ttok, D_MODEL, 0, stream>>>(mF, hF, hB, lnw_l, lnb_l);
    }

    // pool + head
    k_pool<<<Bsz, D_MODEL, 0, stream>>>(hF, lengths, pooled);
    k_head<<<Bsz, 32, 0, stream>>>(pooled, head_w, head_b, out);
}
